// Checkerboard_Window_Attention_60241211294209
// MI455X (gfx1250) — matrix operand order repaired and checked by reading, not compiled, not run on hardware
//
#include <hip/hip_runtime.h>
#include <hip/hip_bf16.h>

typedef _Float16 v4h  __attribute__((ext_vector_type(4)));
typedef _Float16 v8h  __attribute__((ext_vector_type(8)));
typedef _Float16 v16h __attribute__((ext_vector_type(16)));
typedef float    v8f  __attribute__((ext_vector_type(8)));

// 16-byte payload type matching the async-LDS builtin's parameter type
typedef int b128i __attribute__((vector_size(16)));
typedef __attribute__((address_space(1))) b128i* gbl_b128;
typedef __attribute__((address_space(3))) b128i* lds_b128;

#define WIN_N   64
#define DIM     192
#define NHEAD   6
#define DH      32
#define LOGMAX  4.6051702f

// LDS layout (bytes), total 120320 (117.5KB; CDNA5 WGP has 320KB LDS -> 2 WGs/WGP)
#define OFF_XH   0        // f16 [64][200]  = 25600
#define OFF_OH   25600    // f16 [64][200]  = 25600
#define OFF_QH   51200    // f16 [64][32]   = 4096   (overlaps P, dead by then)
#define OFF_KH   55296    // f16 [64][32]   = 4096
#define OFF_P    51200    // f16 [64][72]   = 9216
#define OFF_S    60416    // f32 [64][68]   = 17408
#define OFF_VT   77824    // f16 [32][64]   = 4096
#define OFF_WSL  81920    // f16 [96][200]  = 38400  (per-head qkv weight slice)
#define XSTR 200
#define OSTR 200
#define SSTR 68
#define PSTR 72
#define WSTR 200

// d_ws layout (bytes)
#define WS_WQ    0                       // [6][96][192] f16 = 221184
#define WS_WP    221184                  // [192][192] f16   = 73728
#define WS_BIAS  294912                  // [6][64][64] f32  = 98304

// ---------------- CDNA5 async global->LDS copy (ASYNCcnt-tracked) ----------------
__device__ __forceinline__ void async_g2l_16B(const _Float16* gsrc, _Float16* ldst) {
#if __has_builtin(__builtin_amdgcn_global_load_async_to_lds_b128)
    __builtin_amdgcn_global_load_async_to_lds_b128(
        (gbl_b128)(void*)gsrc, (lds_b128)(void*)ldst, 0, 0);
#else
    unsigned loff = (unsigned)(unsigned long long)
        (__attribute__((address_space(3))) char*)(void*)ldst;
    asm volatile("global_load_async_to_lds_b128 %0, %1, off"
                 :: "v"(loff), "v"(gsrc) : "memory");
#endif
}

__device__ __forceinline__ void wait_async0() {
#if __has_builtin(__builtin_amdgcn_s_wait_asynccnt)
    __builtin_amdgcn_s_wait_asynccnt(0);
#else
    asm volatile("s_wait_asynccnt 0x0" ::: "memory");
#endif
}

// A-fragment (16x32 f16): lane L: row = L%16; K chunks {0..7,16..23} (low half) /
// {8..15,24..31} (high half) -> two 16B loads.
__device__ __forceinline__ v16h load_afrag(const _Float16* __restrict__ base, int stride,
                                           int row, int k0, int lhalf) {
    const _Float16* p = base + row * stride + k0 + lhalf * 8;
    v8h lo = *(const v8h*)p;
    v8h hi = *(const v8h*)(p + 16);
    v16h r;
#pragma unroll
    for (int i = 0; i < 8; ++i) { r[i] = lo[i]; r[i + 8] = hi[i]; }
    return r;
}

// B-fragment (32x16 f16), loaded from row-major B^T: lane L: col N = L%16;
// K chunks {0..7,16..23} (low half) / {8..15,24..31} (high half).
__device__ __forceinline__ v16h load_bfrag(const _Float16* __restrict__ base, int stride,
                                           int nrow, int k0, int lhalf) {
    const _Float16* p = base + nrow * stride + k0 + lhalf * 8;
    v8h lo = *(const v8h*)p;
    v8h hi = *(const v8h*)(p + 16);
    v16h r;
#pragma unroll
    for (int i = 0; i < 8; ++i) { r[i] = lo[i]; r[i + 8] = hi[i]; }
    return r;
}

__device__ __forceinline__ v8f wmma_f16(v16h a, v16h b, v8f c) {
    return __builtin_amdgcn_wmma_f32_16x16x32_f16(false, a, false, b, (short)0, c, false, false);
}

// Stage head h's [96][192] weight slice (rows 0-31 q, 32-63 k, 64-95 v) into LDS.
// wq_perm is pre-permuted so each head slice is one contiguous 36864B region.
__device__ __forceinline__ void issue_wslice_copy(const _Float16* __restrict__ wq_perm,
                                                  _Float16* wsl, int h, int tid) {
    const _Float16* src = wq_perm + (size_t)h * (96 * 192);
#pragma unroll
    for (int j = 0; j < 9; ++j) {
        const int c = tid + j * 256;            // 2304 x 16B chunks
        const int row = c / 24, col = c % 24;
        async_g2l_16B(src + c * 8, wsl + row * WSTR + col * 8);
    }
}

// Pre-pass: f32->f16 weight conversion (qkv permuted per-head) + resolve the
// relative-position bias gather once into a dense [6][64][64] f32 table.
__global__ void prep_kernel(const float* __restrict__ qkv_w,
                            const float* __restrict__ proj_w,
                            const float* __restrict__ bias_table,  // [225][6]
                            const int*   __restrict__ rel_idx,     // [64][64]
                            _Float16* __restrict__ wq_perm,
                            _Float16* __restrict__ wp,
                            float* __restrict__ bias_pre) {
    const int i = blockIdx.x * 256 + threadIdx.x;
    if (i < NHEAD * 96 * 192) {
        const int h = i / (96 * 192);
        const int rem = i % (96 * 192);
        const int r = rem / 192, c = rem % 192;
        const int o = (r >> 5) * 192 + h * 32 + (r & 31);   // q/k/v row in [576]
        wq_perm[i] = (_Float16)qkv_w[o * 192 + c];
        return;
    }
    const int j = i - NHEAD * 96 * 192;
    if (j < 192 * 192) {
        wp[j] = (_Float16)proj_w[j];
        return;
    }
    const int k = j - 192 * 192;
    if (k < NHEAD * 64 * 64) {
        const int h = k / 4096, rc = k % 4096;
        bias_pre[k] = bias_table[rel_idx[rc] * NHEAD + h];
    }
}

__global__ __launch_bounds__(256)
void cb_window_attn_kernel(const float* __restrict__ x,
                           const float* __restrict__ qkv_b,
                           const float* __restrict__ proj_b,
                           const float* __restrict__ logit_scale,
                           const float* __restrict__ bias_pre,     // [6][64][64] f32
                           const _Float16* __restrict__ wq_perm,   // [6][96][192] f16
                           const _Float16* __restrict__ wp,        // [192][192] f16
                           float* __restrict__ out) {
    __shared__ __align__(16) unsigned char smem[120320];
    _Float16* xh  = (_Float16*)(smem + OFF_XH);
    _Float16* oh  = (_Float16*)(smem + OFF_OH);
    _Float16* qh  = (_Float16*)(smem + OFF_QH);
    _Float16* kh  = (_Float16*)(smem + OFF_KH);
    _Float16* Pm  = (_Float16*)(smem + OFF_P);
    float*    Sm  = (float*)   (smem + OFF_S);
    _Float16* vT  = (_Float16*)(smem + OFF_VT);
    _Float16* wsl = (_Float16*)(smem + OFF_WSL);

    const int tid   = threadIdx.x;
    const int wave  = tid >> 5;
    const int lane  = tid & 31;
    const int l16   = lane & 15;
    const int lhalf = lane >> 4;
    const int b     = blockIdx.x;

    // kick off head-0 weight slice copy first so it overlaps x staging
    issue_wslice_copy(wq_perm, wsl, 0, tid);

    // ---- stage x [64,192] f32 -> f16 LDS ----
    const float* xb = x + (size_t)b * (WIN_N * DIM);
#pragma unroll
    for (int i = tid; i < (WIN_N * DIM) / 4; i += 256) {
        float4 v = ((const float4*)xb)[i];
        int e = i * 4, row = e / DIM, col = e % DIM;
        v4h hv; hv[0] = (_Float16)v.x; hv[1] = (_Float16)v.y;
        hv[2] = (_Float16)v.z; hv[3] = (_Float16)v.w;
        *(v4h*)(xh + row * XSTR + col) = hv;
    }
    wait_async0();
    __syncthreads();

    const int m = wave & 3;           // M-tile owned by this wave
    const int rowA = m * 16 + l16;    // A-fragment row
    const bool isQ = wave < 4;

    for (int h = 0; h < NHEAD; ++h) {
        // ===== GEMM1: QKV for this head (B fragments from LDS weight slice) =====
        // waves 0-3: q (2 N-tiles) + v tile0 ; waves 4-7: k (2 N-tiles) + v tile1
        const int vtile  = isQ ? 0 : 1;
        const int qkbase = isQ ? 0 : 32;              // slice rows: q=0..31, k=32..63
        const int o0 = (isQ ? 0 : DIM) + h * DH;      // bias col base for q/k
        const int ov = 2 * DIM + h * DH + vtile * 16; // bias col base for v
        v8f acc0 = {0,0,0,0,0,0,0,0};
        v8f acc1 = {0,0,0,0,0,0,0,0};
        v8f accv = {0,0,0,0,0,0,0,0};
#pragma unroll
        for (int kk = 0; kk < 6; ++kk) {
            const int k0 = kk * 32;
            v16h a  = load_afrag(xh, XSTR, rowA, k0, lhalf);
            v16h b0 = load_bfrag(wsl, WSTR, qkbase + l16,              k0, lhalf);
            v16h b1 = load_bfrag(wsl, WSTR, qkbase + 16 + l16,         k0, lhalf);
            v16h bv = load_bfrag(wsl, WSTR, 64 + vtile * 16 + l16,     k0, lhalf);
            acc0 = wmma_f16(a, b0, acc0);
            acc1 = wmma_f16(a, b1, acc1);
            accv = wmma_f16(a, bv, accv);
        }
        const float bq0 = qkv_b[o0 + l16];
        const float bq1 = qkv_b[o0 + 16 + l16];
        const float bvv = qkv_b[ov + l16];
        _Float16* dst = isQ ? qh : kh;
#pragma unroll
        for (int r = 0; r < 8; ++r) {
            float a0 = acc0[r] + bq0;
            float a1 = acc1[r] + bq1;
            float av = accv[r] + bvv;
            const int tok = m * 16 + r + 8 * lhalf;
            // v stored transposed: vT[d][token]
            vT[(vtile * 16 + l16) * 64 + tok] = (_Float16)av;
            // per-row L2 norm of q/k across 32 dims (2 tiles, 16 lanes of this half)
            float ss = a0 * a0 + a1 * a1;
            ss += __shfl_xor(ss, 1); ss += __shfl_xor(ss, 2);
            ss += __shfl_xor(ss, 4); ss += __shfl_xor(ss, 8);
            const float inv = 1.0f / fmaxf(sqrtf(ss), 1e-12f);
            dst[tok * DH + l16]      = (_Float16)(a0 * inv);
            dst[tok * DH + 16 + l16] = (_Float16)(a1 * inv);
        }
        __syncthreads();   // q/k/v staged; weight slice fully consumed

        // prefetch next head's weight slice (overlaps S/softmax/PV phases)
        if (h + 1 < NHEAD) issue_wslice_copy(wq_perm, wsl, h + 1, tid);

        // ===== S = qn @ kn^T : one 16x16x32 WMMA per tile (K=DH=32) =====
        {
            const int sm = wave & 3, nb = (wave >> 2) * 2;
            v16h aq  = load_afrag(qh, DH, sm * 16 + l16, 0, lhalf);
            v16h bk0 = load_bfrag(kh, DH, nb * 16 + l16,       0, lhalf);
            v16h bk1 = load_bfrag(kh, DH, (nb + 1) * 16 + l16, 0, lhalf);
            v8f s0 = {0,0,0,0,0,0,0,0};
            v8f s1 = {0,0,0,0,0,0,0,0};
            s0 = wmma_f16(aq, bk0, s0);
            s1 = wmma_f16(aq, bk1, s1);
#pragma unroll
            for (int r = 0; r < 8; ++r) {
                const int rr = sm * 16 + r + 8 * lhalf;
                Sm[rr * SSTR + nb * 16 + l16]       = s0[r];
                Sm[rr * SSTR + (nb + 1) * 16 + l16] = s1[r];
            }
        }
        __syncthreads();   // S complete

        // ===== scale + rel-pos bias + checkerboard mask + softmax -> P f16 =====
        {
            const float scale = __expf(fminf(logit_scale[h], LOGMAX));
            const int r = tid >> 2, qd = tid & 3, cbase = qd * 16;
            const int fr = ((r >> 3) + (r & 7)) & 1;   // 0 => row kept
            // coalesced bias row segment: 4 x float4 (precomputed gather)
            const float4* bp = (const float4*)(bias_pre + h * 4096 + r * 64 + cbase);
            float bias[16];
#pragma unroll
            for (int c4 = 0; c4 < 4; ++c4) {
                float4 bv4 = bp[c4];
                bias[c4 * 4 + 0] = bv4.x; bias[c4 * 4 + 1] = bv4.y;
                bias[c4 * 4 + 2] = bv4.z; bias[c4 * 4 + 3] = bv4.w;
            }
            float vals[16];
            float mx = -3.0e38f;
#pragma unroll
            for (int c = 0; c < 16; ++c) {
                const int cc = cbase + c;
                const int fc = ((cc >> 3) + (cc & 7)) & 1;
                const bool keep = (fr == 0) && (fc == 0);
                const float vv = Sm[r * SSTR + cc] * scale + bias[c];
                vals[c] = vv;
                if (keep) mx = fmaxf(mx, vv);
            }
            mx = fmaxf(mx, __shfl_xor(mx, 1));
            mx = fmaxf(mx, __shfl_xor(mx, 2));
            const float msafe = (mx > -1.0e37f) ? mx : 0.0f;
            float sum = 0.0f;
            float ev[16];
#pragma unroll
            for (int c = 0; c < 16; ++c) {
                const int cc = cbase + c;
                const int fc = ((cc >> 3) + (cc & 7)) & 1;
                const bool keep = (fr == 0) && (fc == 0);
                const float e = keep ? __expf(vals[c] - msafe) : 0.0f;
                ev[c] = e; sum += e;
            }
            sum += __shfl_xor(sum, 1);
            sum += __shfl_xor(sum, 2);
            const float rinv = (sum > 0.0f) ? (1.0f / sum) : 0.0f;
            // two 16B packed stores (row segment is contiguous & 16B aligned)
            v8h p0, p1;
#pragma unroll
            for (int c = 0; c < 8; ++c) {
                p0[c] = (_Float16)(ev[c] * rinv);
                p1[c] = (_Float16)(ev[c + 8] * rinv);
            }
            *(v8h*)(Pm + r * PSTR + cbase)     = p0;
            *(v8h*)(Pm + r * PSTR + cbase + 8) = p1;
        }
        __syncthreads();   // P ready (q/k region consumed)

        // ===== GEMM2: O_h = P @ V  (one 16x16 tile per wave, K=64) =====
        {
            const int m2 = wave & 3, n2 = wave >> 2;
            v8f o = {0,0,0,0,0,0,0,0};
#pragma unroll
            for (int kk = 0; kk < 2; ++kk) {
                v16h ap = load_afrag(Pm, PSTR, m2 * 16 + l16, kk * 32, lhalf);
                v16h bv = load_bfrag(vT, 64,   n2 * 16 + l16, kk * 32, lhalf);
                o = wmma_f16(ap, bv, o);
            }
#pragma unroll
            for (int r = 0; r < 8; ++r)
                oh[(m2 * 16 + r + 8 * lhalf) * OSTR + h * DH + n2 * 16 + l16] = (_Float16)o[r];
        }
        wait_async0();     // next head's weight slice landed
        __syncthreads();   // oh slice done; P/vT/wsl reusable next head
    }

    // ===== proj: y = oh[64,192] @ proj_w^T + proj_b -> fp32 out =====
    {
        const int nb = (wave >> 2) * 6;
        float* ob = out + (size_t)b * (WIN_N * DIM);
#pragma unroll
        for (int nt = 0; nt < 6; ++nt) {
            const int n = nb + nt;
            v8f acc = {0,0,0,0,0,0,0,0};
#pragma unroll
            for (int kk = 0; kk < 6; ++kk) {
                v16h a  = load_afrag(oh, OSTR, rowA, kk * 32, lhalf);
                v16h bw = load_bfrag(wp, DIM, n * 16 + l16, kk * 32, lhalf);
                acc = wmma_f16(a, bw, acc);
            }
            const float pb = proj_b[n * 16 + l16];
#pragma unroll
            for (int r = 0; r < 8; ++r)
                ob[(m * 16 + r + 8 * lhalf) * DIM + n * 16 + l16] = acc[r] + pb;
        }
    }
}

extern "C" void kernel_launch(void* const* d_in, const int* in_sizes, int n_in,
                              void* d_out, int out_size, void* d_ws, size_t ws_size,
                              hipStream_t stream) {
    const float* x           = (const float*)d_in[0];
    const float* qkv_w       = (const float*)d_in[1];
    const float* qkv_b       = (const float*)d_in[2];
    const float* proj_w      = (const float*)d_in[3];
    const float* proj_b      = (const float*)d_in[4];
    const float* logit_scale = (const float*)d_in[5];
    const float* bias_table  = (const float*)d_in[6];
    const int*   rel_idx     = (const int*)d_in[7];

    _Float16* wq_perm  = (_Float16*)((char*)d_ws + WS_WQ);
    _Float16* wp       = (_Float16*)((char*)d_ws + WS_WP);
    float*    bias_pre = (float*)   ((char*)d_ws + WS_BIAS);

    const int prep_items = NHEAD * 96 * 192 + 192 * 192 + NHEAD * 64 * 64;
    prep_kernel<<<(prep_items + 255) / 256, 256, 0, stream>>>(
        qkv_w, proj_w, bias_table, rel_idx, wq_perm, wp, bias_pre);

    cb_window_attn_kernel<<<8192, 256, 0, stream>>>(
        x, qkv_b, proj_b, logit_scale, bias_pre, wq_perm, wp, (float*)d_out);
}
